// RoIPooling_49168785604639
// MI455X (gfx1250) — matrix operand order repaired and checked by reading, not compiled, not run on hardware
//
#include <hip/hip_runtime.h>
#include <stdint.h>

typedef __attribute__((ext_vector_type(16))) __bf16 v16bf;
typedef __attribute__((ext_vector_type(8)))  float  v8f;

#define R_NUM   1024
#define CCH     1024
#define FHW     19
#define FSP     (FHW*FHW)
#define PSP     25
#define EPSV    1e-5f
#define ROISC   (300.0f/16.0f)

// ---------------- workspace layout (bytes) ----------------
static constexpr size_t OFF_WY   = 0;
static constexpr size_t SZ_W     = (size_t)R_NUM * 95 * 4;            // 5x19 f32 per roi
static constexpr size_t OFF_WX   = OFF_WY + SZ_W;
static constexpr size_t OFF_POOL = OFF_WX + SZ_W;                      // bf16 [r][25][1024]
static constexpr size_t SZ_POOL  = (size_t)R_NUM * PSP * CCH * 2;
static constexpr size_t OFF_A1   = OFF_POOL + SZ_POOL;                 // bf16 frag [288][16][32][16]
static constexpr size_t SZ_A1    = (size_t)288 * 16 * 512 * 2;
static constexpr size_t OFF_H1   = OFF_A1 + SZ_A1;                     // bf16 [r][9][256]
static constexpr size_t SZ_H1    = (size_t)R_NUM * 9 * 256 * 2;
static constexpr size_t OFF_A2   = OFF_H1 + SZ_H1;                     // bf16 frag [72][16][32][16]

__device__ __forceinline__ int clampi(int v, int lo, int hi) {
    return v < lo ? lo : (v > hi ? hi : v);
}

// ------------- 1. per-RoI adaptive pooling weights -------------
__global__ void roi_weights_kernel(const float* __restrict__ rois,
                                   float* __restrict__ wy, float* __restrict__ wx) {
    int r = blockIdx.x * blockDim.x + threadIdx.x;
    if (r >= R_NUM) return;
    const float* rr = rois + (size_t)r * 5;
    int x0 = clampi((int)(rr[1] * ROISC), 1, FHW - 1);
    int y0 = clampi((int)(rr[2] * ROISC), 1, FHW - 1);
    int x1 = clampi((int)(rr[3] * ROISC), 1, FHW - 1);
    int y1 = clampi((int)(rr[4] * ROISC), 1, FHW - 1);
    if (x1 == x0) x0 -= 1;
    if (y1 == y0) y0 -= 1;
    int nx = x1 - x0 + 1;
    int ny = y1 - y0 + 1;
    for (int i = 0; i < 5; ++i) {
        int sy = y0 + (i * ny) / 5;
        int ey = y0 + ((i + 1) * ny + 4) / 5;
        float wv = 1.0f / (float)(ey - sy);
        for (int y = 0; y < FHW; ++y)
            wy[((size_t)r * 5 + i) * FHW + y] = (y >= sy && y < ey) ? wv : 0.0f;
        int sx = x0 + (i * nx) / 5;
        int ex = x0 + ((i + 1) * nx + 4) / 5;
        float wh = 1.0f / (float)(ex - sx);
        for (int x = 0; x < FHW; ++x)
            wx[((size_t)r * 5 + i) * FHW + x] = (x >= sx && x < ex) ? wh : 0.0f;
    }
}

// ------------- 2. separable RoI pooling -> bf16, transposed [r][sp][ic] -------------
__global__ void __launch_bounds__(256) pool_kernel(const float* __restrict__ feat,
                                                   const float* __restrict__ wy,
                                                   const float* __restrict__ wx,
                                                   __bf16* __restrict__ pooled) {
    __shared__ float swy[95];
    __shared__ float swx[95];
    int r   = blockIdx.x;
    int tid = threadIdx.x;
    if (tid < 95) {
        swy[tid] = wy[(size_t)r * 95 + tid];
        swx[tid] = wx[(size_t)r * 95 + tid];
    }
    __syncthreads();
    int c = blockIdx.y * 256 + tid;
    const float* f = feat + (size_t)c * FSP;
    float acc[5][5];
#pragma unroll
    for (int i = 0; i < 5; ++i)
#pragma unroll
        for (int j = 0; j < 5; ++j) acc[i][j] = 0.0f;

    for (int y = 0; y < FHW; ++y) {
        float row[5] = {0.f, 0.f, 0.f, 0.f, 0.f};
        for (int x = 0; x < FHW; ++x) {
            float fv = f[y * FHW + x];
#pragma unroll
            for (int j = 0; j < 5; ++j) row[j] += swx[j * FHW + x] * fv;
        }
#pragma unroll
        for (int i = 0; i < 5; ++i) {
            float wv = swy[i * FHW + y];
#pragma unroll
            for (int j = 0; j < 5; ++j) acc[i][j] += wv * row[j];
        }
    }
    __bf16* po = pooled + (size_t)r * PSP * CCH;
#pragma unroll
    for (int i = 0; i < 5; ++i)
#pragma unroll
        for (int j = 0; j < 5; ++j)
            po[(size_t)(i * 5 + j) * CCH + c] = (__bf16)acc[i][j];
}

// ------------- 3. weight prepack into WMMA A-fragment order -------------
// A 16-bit 16x32 layout: lane(0..15)=M, K = khalf*16+e (contiguous, matches B fill)
__device__ __forceinline__ int a_klocal(int lane, int e) {
    int khalf = lane >> 4;
    return e + khalf * 16;
}

__global__ void pack_a1_kernel(const float* __restrict__ w1, __bf16* __restrict__ a1) {
    int idx = blockIdx.x * 256 + threadIdx.x;         // 288*16*32*16 = 2359296
    int e     = idx & 15;
    int lane  = (idx >> 4) & 31;
    int mtile = (idx >> 9) & 15;
    int kk    = idx >> 13;                             // 0..287
    int kl    = a_klocal(lane, e);
    int kpos  = kk >> 5;                               // 0..8 (ky*3+kx)
    int ic    = (kk & 31) * 32 + kl;                   // 0..1023
    int m     = mtile * 16 + (lane & 15);              // oc
    a1[idx] = (__bf16)w1[((size_t)m * CCH + ic) * 9 + kpos];
}

__global__ void pack_a2_kernel(const float* __restrict__ w2, __bf16* __restrict__ a2) {
    int idx = blockIdx.x * 256 + threadIdx.x;         // 72*16*32*16 = 589824
    int e     = idx & 15;
    int lane  = (idx >> 4) & 31;
    int mtile = (idx >> 9) & 15;
    int kk    = idx >> 13;                             // 0..71
    int kl    = a_klocal(lane, e);
    int kpos  = kk >> 3;                               // 0..8
    int ic    = (kk & 7) * 32 + kl;                    // 0..255
    int m     = mtile * 16 + (lane & 15);
    a2[idx] = (__bf16)w2[((size_t)m * 256 + ic) * 9 + kpos];
}

// ------------- 4. conv1 (1024->256, 5x5 -> 3x3, stride 2 pad 1) + BN + ReLU -------------
// LDS: [25][1024] activation tile + one zero-filled 1024-half guard stripe for padding taps.
__global__ void __launch_bounds__(256) conv1_wmma_kernel(
    const __bf16* __restrict__ pooled, const __bf16* __restrict__ a1,
    const float* __restrict__ b1, const float* __restrict__ g1,
    const float* __restrict__ be1, const float* __restrict__ m1,
    const float* __restrict__ v1, __bf16* __restrict__ h1) {
    __shared__ __bf16 sp[(PSP + 1) * CCH];            // 53248 B
    int r   = blockIdx.x;
    int tid = threadIdx.x;

    // stage pooled tile for this RoI (12800 dwords, 50 per thread) + zero guard stripe
    const uint32_t* gp = (const uint32_t*)(pooled + (size_t)r * PSP * CCH);
    uint32_t* lp = (uint32_t*)sp;
#pragma unroll 5
    for (int i = tid; i < PSP * CCH / 2; i += 256) lp[i] = gp[i];
#pragma unroll
    for (int i = tid; i < CCH / 2; i += 256) lp[PSP * CCH / 2 + i] = 0u;
    __syncthreads();

    int lane = tid & 31;
    int wave = tid >> 5;
    int mt0  = wave * 2;
    int mt1  = wave * 2 + 1;
    int n    = lane & 15;                              // output column (oy*3+ox), 9 valid
    int oy   = n / 3, ox = n % 3;
    int kofs = (lane >> 4) << 4;                       // B: lanes 16-31 hold K+16

    v8f acc0 = {};
    v8f acc1 = {};
    const v16bf* afr = (const v16bf*)a1;
    const v16bf* bls = (const v16bf*)sp;

    for (int kpos = 0; kpos < 9; ++kpos) {
        int ky = kpos / 3, kx = kpos % 3;
        int iy = 2 * oy - 1 + ky;
        int ix = 2 * ox - 1 + kx;
        bool valid = (n < 9) & (iy >= 0) & (iy < 5) & (ix >= 0) & (ix < 5);
        // invalid lanes read the zero guard stripe: no per-iteration data select
        int base = (valid ? (iy * 5 + ix) : PSP) * CCH + kofs;
        const v16bf* afr_k = afr + ((size_t)kpos * 32 * 16) * 32;   // 32 k-blocks per tap
#pragma unroll 4
        for (int kb = 0; kb < 32; ++kb) {
            v16bf bf  = bls[(base + kb * 32) >> 4];
            v16bf av0 = afr_k[((size_t)kb * 16 + mt0) * 32 + lane];
            v16bf av1 = afr_k[((size_t)kb * 16 + mt1) * 32 + lane];
            acc0 = __builtin_amdgcn_wmma_f32_16x16x32_bf16(false, av0, false, bf,
                                                           (short)0, acc0, false, false);
            acc1 = __builtin_amdgcn_wmma_f32_16x16x32_bf16(false, av1, false, bf,
                                                           (short)0, acc1, false, false);
        }
    }

    // epilogue: bias + BN + ReLU, write bf16 transposed [r][pos][oc]
    if (n < 9) {
        int mrow = (lane >= 16) ? 8 : 0;
#pragma unroll
        for (int i = 0; i < 8; ++i) {
            {
                int oc = mt0 * 16 + mrow + i;
                float sc = g1[oc] * rsqrtf(v1[oc] + EPSV);
                float sh = be1[oc] - m1[oc] * sc;
                float y  = (acc0[i] + b1[oc]) * sc + sh;
                h1[((size_t)r * 9 + n) * 256 + oc] = (__bf16)fmaxf(y, 0.0f);
            }
            {
                int oc = mt1 * 16 + mrow + i;
                float sc = g1[oc] * rsqrtf(v1[oc] + EPSV);
                float sh = be1[oc] - m1[oc] * sc;
                float y  = (acc1[i] + b1[oc]) * sc + sh;
                h1[((size_t)r * 9 + n) * 256 + oc] = (__bf16)fmaxf(y, 0.0f);
            }
        }
    }
}

// ------------- 5. conv2 (256->256, 3x3 -> 2x2, stride 2 pad 1) + BN + ReLU -> out -------------
__global__ void __launch_bounds__(256) conv2_wmma_kernel(
    const __bf16* __restrict__ h1, const __bf16* __restrict__ a2,
    const float* __restrict__ b2, const float* __restrict__ g2,
    const float* __restrict__ be2, const float* __restrict__ m2,
    const float* __restrict__ v2, float* __restrict__ out) {
    __shared__ __bf16 sh1[10 * 256];                   // 5120 B incl. zero guard stripe
    int r   = blockIdx.x;
    int tid = threadIdx.x;

    const uint32_t* gp = (const uint32_t*)(h1 + (size_t)r * 9 * 256);
    uint32_t* lp = (uint32_t*)sh1;
    for (int i = tid; i < 9 * 256 / 2; i += 256) lp[i] = gp[i];
    if (tid < 128) lp[9 * 128 + tid] = 0u;             // zero guard stripe (256 halves)
    __syncthreads();

    int lane = tid & 31;
    int wave = tid >> 5;
    int mt0  = wave * 2;
    int mt1  = wave * 2 + 1;
    int n    = lane & 15;                              // output column (oy*2+ox), 4 valid
    int oy   = n >> 1, ox = n & 1;
    int kofs = (lane >> 4) << 4;

    v8f acc0 = {};
    v8f acc1 = {};
    const v16bf* afr = (const v16bf*)a2;
    const v16bf* bls = (const v16bf*)sh1;

    for (int kpos = 0; kpos < 9; ++kpos) {
        int ky = kpos / 3, kx = kpos % 3;
        int iy = 2 * oy - 1 + ky;
        int ix = 2 * ox - 1 + kx;
        bool valid = (n < 4) & (iy >= 0) & (iy < 3) & (ix >= 0) & (ix < 3);
        int base = (valid ? (iy * 3 + ix) : 9) * 256 + kofs;
        const v16bf* afr_k = afr + ((size_t)kpos * 8 * 16) * 32;    // 8 k-blocks per tap
#pragma unroll
        for (int kb = 0; kb < 8; ++kb) {
            v16bf bf  = bls[(base + kb * 32) >> 4];
            v16bf av0 = afr_k[((size_t)kb * 16 + mt0) * 32 + lane];
            v16bf av1 = afr_k[((size_t)kb * 16 + mt1) * 32 + lane];
            acc0 = __builtin_amdgcn_wmma_f32_16x16x32_bf16(false, av0, false, bf,
                                                           (short)0, acc0, false, false);
            acc1 = __builtin_amdgcn_wmma_f32_16x16x32_bf16(false, av1, false, bf,
                                                           (short)0, acc1, false, false);
        }
    }

    if (n < 4) {
        int mrow = (lane >= 16) ? 8 : 0;
#pragma unroll
        for (int i = 0; i < 8; ++i) {
            {
                int oc = mt0 * 16 + mrow + i;
                float sc = g2[oc] * rsqrtf(v2[oc] + EPSV);
                float sh = be2[oc] - m2[oc] * sc;
                float y  = (acc0[i] + b2[oc]) * sc + sh;
                out[((size_t)r * 256 + oc) * 4 + n] = fmaxf(y, 0.0f);
            }
            {
                int oc = mt1 * 16 + mrow + i;
                float sc = g2[oc] * rsqrtf(v2[oc] + EPSV);
                float sh = be2[oc] - m2[oc] * sc;
                float y  = (acc1[i] + b2[oc]) * sc + sh;
                out[((size_t)r * 256 + oc) * 4 + n] = fmaxf(y, 0.0f);
            }
        }
    }
}

// ---------------------------------------------------------------
extern "C" void kernel_launch(void* const* d_in, const int* in_sizes, int n_in,
                              void* d_out, int out_size, void* d_ws, size_t ws_size,
                              hipStream_t stream) {
    const float* rois = (const float*)d_in[0];
    const float* feat = (const float*)d_in[1];
    const float* w1   = (const float*)d_in[2];
    const float* b1   = (const float*)d_in[3];
    const float* g1   = (const float*)d_in[4];
    const float* be1  = (const float*)d_in[5];
    const float* m1   = (const float*)d_in[6];
    const float* v1   = (const float*)d_in[7];
    const float* w2   = (const float*)d_in[8];
    const float* b2   = (const float*)d_in[9];
    const float* g2   = (const float*)d_in[10];
    const float* be2  = (const float*)d_in[11];
    const float* m2   = (const float*)d_in[12];
    const float* v2   = (const float*)d_in[13];
    float* out = (float*)d_out;

    char* ws = (char*)d_ws;
    float*  wy     = (float*)(ws + OFF_WY);
    float*  wx     = (float*)(ws + OFF_WX);
    __bf16* pooled = (__bf16*)(ws + OFF_POOL);
    __bf16* a1     = (__bf16*)(ws + OFF_A1);
    __bf16* h1     = (__bf16*)(ws + OFF_H1);
    __bf16* a2     = (__bf16*)(ws + OFF_A2);

    roi_weights_kernel<<<4, 256, 0, stream>>>(rois, wy, wx);
    pack_a1_kernel<<<9216, 256, 0, stream>>>(w1, a1);
    pack_a2_kernel<<<2304, 256, 0, stream>>>(w2, a2);
    pool_kernel<<<dim3(R_NUM, 4), 256, 0, stream>>>(feat, wy, wx, pooled);
    conv1_wmma_kernel<<<R_NUM, 256, 0, stream>>>(pooled, a1, b1, g1, be1, m1, v1, h1);
    conv2_wmma_kernel<<<R_NUM, 256, 0, stream>>>(h1, a2, b2, g2, be2, m2, v2, out);
}
